// BondFlow_9259949490697
// MI455X (gfx1250) — matrix operand order repaired and checked by reading, not compiled, not run on hardware
//
#include <hip/hip_runtime.h>

typedef _Float16 f16;
typedef __attribute__((ext_vector_type(16))) _Float16 v16h;
typedef __attribute__((ext_vector_type(8)))  float    v8f;

#define GF_SIGMOID 1
#define GF_GATE    2
#define GF_BIAS    4

static constexpr int NQn  = 4096;
static constexpr int NCn  = 8192;
static constexpr int KKNN = 24;
static constexpr int DEGQ = 8;
static constexpr int EKn  = NQn * KKNN;   // 98304
static constexpr int EQn  = NQn * DEGQ;   // 32768
static constexpr int Pn   = EQn * DEGQ;   // 262144

// ---------------------------------------------------------------------------
// WMMA GEMM:  C[M,N] = epilogue( [A1|A2][M,K] * W[N,K]^T )
//  - per wave: 32x16 output tile (two 16x16 WMMA accumulators, B reused)
//  - main K loop: unconditional b128 loads (all strides are 8-half multiples)
//  - tail (K%32, only the K=4 gate GEMM) uses guarded 32-bit pair loads
// ---------------------------------------------------------------------------
struct GemmArgs {
  const f16* A1; const f16* A2; const f16* W;
  const int* idxA; const float* bias; const float* gate;
  float* Cf32; f16* Cf16;
  int M, N, K, K1, lda1, lda2, ldw, flags;
};

__launch_bounds__(128)
__global__ void k_gemm(GemmArgs a) {
  const int wave = threadIdx.x >> 5;
  const int lane = threadIdx.x & 31;
  const int m0 = blockIdx.x << 5;                 // 32 rows / block
  const int n0 = (blockIdx.y * 4 + wave) << 4;    // 16 cols / wave
  if (n0 >= a.N) return;                          // wave-uniform exit
  const int g = lane >> 4;
  const int l = lane & 15;
  int mrA = m0 + l;      if (mrA >= a.M) mrA = a.M - 1;
  int mrB = m0 + 16 + l; if (mrB >= a.M) mrB = a.M - 1;
  const int arA = a.idxA ? a.idxA[mrA] : mrA;
  const int arB = a.idxA ? a.idxA[mrB] : mrB;
  const f16* a1pA = a.A1 + (size_t)arA * a.lda1;
  const f16* a1pB = a.A1 + (size_t)arB * a.lda1;
  const f16* a2pA = a.A2 ? (a.A2 + (size_t)arA * a.lda2) : a1pA;
  const f16* a2pB = a.A2 ? (a.A2 + (size_t)arB * a.lda2) : a1pB;
  const int nr = n0 + l;
  const bool nok = nr < a.N;
  const f16* wp = a.W + (size_t)(nok ? nr : 0) * a.ldw;
  v8f acc0 = {}, acc1 = {};
  const int kmain = a.K & ~31;
#pragma unroll 2
  for (int k0 = 0; k0 < kmain; k0 += 32) {
    const int kc1 = k0 + (g << 3);        // A chunk 1: halves [kc1, kc1+8)
    const int kc2 = kc1 + 16;             // A chunk 2
    const int kw = k0 + (g << 3);         // B: halves [kw, kw+8) and [kw+16, kw+24)
    union { v16h v; uint4 q[2]; } fa, fb, fw;
    fa.q[0] = *(const uint4*)((kc1 < a.K1) ? (a1pA + kc1) : (a2pA + (kc1 - a.K1)));
    fa.q[1] = *(const uint4*)((kc2 < a.K1) ? (a1pA + kc2) : (a2pA + (kc2 - a.K1)));
    fb.q[0] = *(const uint4*)((kc1 < a.K1) ? (a1pB + kc1) : (a2pB + (kc1 - a.K1)));
    fb.q[1] = *(const uint4*)((kc2 < a.K1) ? (a1pB + kc2) : (a2pB + (kc2 - a.K1)));
    const f16* wrow = wp + kw;
    fw.q[0] = *(const uint4*)wrow;
    fw.q[1] = *(const uint4*)(wrow + 16);
    __builtin_prefetch((const void*)(wrow + 512), 0, 1);  // global_prefetch_b8
    acc0 = __builtin_amdgcn_wmma_f32_16x16x32_f16(false, fa.v, false, fw.v,
                                                  (short)0, acc0, false, false);
    acc1 = __builtin_amdgcn_wmma_f32_16x16x32_f16(false, fb.v, false, fw.v,
                                                  (short)0, acc1, false, false);
  }
  if (kmain < a.K) {                      // tail (K % 32): guarded pair loads
    union { v16h v; unsigned u[8]; } ta, tb, tw;
#pragma unroll
    for (int r = 0; r < 8; ++r) {
      const int kb = (r < 4) ? (8 * g + 2 * r) : (16 + 8 * g + 2 * (r - 4));
      const int k = kmain + kb;
      unsigned pa = 0u, pb = 0u, pw = 0u;
      if (k < a.K) {
        pa = *(const unsigned*)((k < a.K1) ? (a1pA + k) : (a2pA + (k - a.K1)));
        pb = *(const unsigned*)((k < a.K1) ? (a1pB + k) : (a2pB + (k - a.K1)));
      }
      const int kw = kmain + ((r < 4) ? (8 * g + 2 * r) : (16 + 8 * g + 2 * (r - 4)));
      if (nok && kw < a.K) pw = *(const unsigned*)(wp + kw);
      ta.u[r] = pa; tb.u[r] = pb; tw.u[r] = pw;
    }
    acc0 = __builtin_amdgcn_wmma_f32_16x16x32_f16(false, ta.v, false, tw.v,
                                                  (short)0, acc0, false, false);
    acc1 = __builtin_amdgcn_wmma_f32_16x16x32_f16(false, tb.v, false, tw.v,
                                                  (short)0, acc1, false, false);
  }
  const float bn = ((a.flags & GF_BIAS) && nok) ? a.bias[nr] : 0.f;
#pragma unroll
  for (int r = 0; r < 8; ++r) {
    for (int t = 0; t < 2; ++t) {
      const int m = m0 + (t << 4) + r + (g << 3);
      if (!nok || m >= a.M) continue;
      float v = (t ? acc1[r] : acc0[r]) + bn;
      if (a.flags & GF_SIGMOID) v = 1.f / (1.f + __expf(-v));
      if (a.flags & GF_GATE) v *= a.gate[(size_t)m * a.N + nr];
      const size_t o = (size_t)m * a.N + nr;
      if (a.Cf32) a.Cf32[o] = v;
      if (a.Cf16) a.Cf16[o] = (f16)v;
    }
  }
}

// ---------------------------------------------------------------------------
// elementwise / graph kernels
// ---------------------------------------------------------------------------
__global__ void k_cvt(const float* s, f16* d, long long n) {
  long long i = (long long)blockIdx.x * blockDim.x + threadIdx.x;
  if (i < n) d[i] = (f16)s[i];
}
__global__ void k_planes(const float* s, f16* p0, f16* p1, f16* p2, long long n) {
  long long i = (long long)blockIdx.x * blockDim.x + threadIdx.x;
  if (i >= n) return;
  p0[i] = (f16)s[3 * i]; p1[i] = (f16)s[3 * i + 1]; p2[i] = (f16)s[3 * i + 2];
}
__global__ void k_vecnorm3(const f16* a, const f16* b, const f16* c, f16* o, long long n) {
  long long i = (long long)blockIdx.x * blockDim.x + threadIdx.x;
  if (i >= n) return;
  float x = (float)a[i], y = (float)b[i], z = (float)c[i];
  o[i] = (f16)sqrtf(x * x + y * y + z * z + 1e-12f);
}
__global__ void k_leaky(const float* s, f16* d, long long n) {
  long long i = (long long)blockIdx.x * blockDim.x + threadIdx.x;
  if (i >= n) return;
  float v = s[i]; d[i] = (f16)(v < 0.f ? 0.01f * v : v);
}
__global__ void k_vnleaky(f16* v0, f16* v1, f16* v2,
                          const f16* d0, const f16* d1, const f16* d2, long long n) {
  long long i = (long long)blockIdx.x * blockDim.x + threadIdx.x;
  if (i >= n) return;
  float x0 = (float)v0[i], x1 = (float)v1[i], x2 = (float)v2[i];
  float y0 = (float)d0[i], y1 = (float)d1[i], y2 = (float)d2[i];
  float dot = x0 * y0 + x1 * y1 + x2 * y2;
  float dd = y0 * y0 + y1 * y1 + y2 * y2;
  float o0 = x0, o1 = x1, o2 = x2;
  if (dot < 0.f) {
    float f = dot / (dd + 1e-6f);
    o0 = x0 - f * y0; o1 = x1 - f * y1; o2 = x2 - f * y2;
  }
  v0[i] = (f16)(0.2f * x0 + 0.8f * o0);
  v1[i] = (f16)(0.2f * x1 + 0.8f * o1);
  v2[i] = (f16)(0.2f * x2 + 0.8f * o2);
}
// dist -> gaussian smearing (+extra cols) and unit-vec * vexp weights
__global__ void k_edge_geom(const float* posA, const float* posB,
                            const int* ia, const int* ib, int E,
                            const float* vw, int C,
                            f16* sca, int ldsca, float stop, int ng,
                            const float* extra, int nextra,
                            f16* w0, f16* w1, f16* w2, int ldv) {
  int e = blockIdx.x * blockDim.x + threadIdx.x;
  if (e >= E) return;
  int ra = ia ? ia[e] : e;
  float x = posA[3 * ra], y = posA[3 * ra + 1], z = posA[3 * ra + 2];
  if (posB) { int rb = ib[e]; x -= posB[3 * rb]; y -= posB[3 * rb + 1]; z -= posB[3 * rb + 2]; }
  float d = sqrtf(x * x + y * y + z * z);
  float inv = 1.f / (d + 1e-7f);
  float ux = x * inv, uy = y * inv, uz = z * inv;
  if (sca && ng > 0) {
    float step = stop / (float)(ng - 1);
    float coeff = -0.5f / (step * step);
    for (int i = 0; i < ng; ++i) {
      float t = d - step * (float)i;
      sca[(size_t)e * ldsca + i] = (f16)__expf(coeff * t * t);
    }
    for (int i = 0; i < nextra; ++i)
      sca[(size_t)e * ldsca + ng + i] = (f16)extra[(size_t)e * nextra + i];
  }
  for (int c = 0; c < C; ++c) {
    float w = vw[c];
    w0[(size_t)e * ldv + c] = (f16)(w * ux);
    w1[(size_t)e * ldv + c] = (f16)(w * uy);
    w2[(size_t)e * ldv + c] = (f16)(w * uz);
  }
}
__global__ void k_mulg(const f16* nsrc, const int* idx, const f16* t, f16* y,
                       int cols, long long n) {
  long long i = (long long)blockIdx.x * blockDim.x + threadIdx.x;
  if (i >= n) return;
  int r = (int)(i / cols), c = (int)(i % cols);
  y[i] = (f16)((float)nsrc[(size_t)idx[r] * cols + c] * (float)t[i]);
}
__global__ void k_yvec(const float* cA, const float* cB, const int* cj,
                       const f16* n0, const f16* n1, const f16* n2,
                       const f16* e0, const f16* e1, const f16* e2,
                       f16* y0, f16* y1, f16* y2, long long n) {
  long long i = (long long)blockIdx.x * blockDim.x + threadIdx.x;
  if (i >= n) return;
  int e = (int)(i >> 6), o = (int)(i & 63);
  size_t ni = (size_t)cj[e] * 64 + o;
  float a = cA[i], b = cB[i];
  y0[i] = (f16)(a * (float)n0[ni] + b * (float)e0[i]);
  y1[i] = (f16)(a * (float)n1[ni] + b * (float)e1[i]);
  y2[i] = (f16)(a * (float)n2[ni] + b * (float)e2[i]);
}
__global__ void k_att1(const int* qi, const f16* qs, const f16* ks,
                       const f16* q0, const f16* q1, const f16* q2,
                       const f16* k0, const f16* k1, const f16* k2,
                       float* sl, float* vl, int E) {
  int e = blockIdx.x * blockDim.x + threadIdx.x;
  if (e >= E) return;
  int q = qi[e];
  const f16* a = qs + (size_t)q * 256;
  const f16* b = ks + (size_t)e * 256;
  float s = 0.f;
  for (int d = 0; d < 256; ++d) s += (float)a[d] * (float)b[d];
  sl[e] = s * (1.f / 16.f);
  float v = 0.f;
  size_t qo = (size_t)q * 64, eo = (size_t)e * 64;
  for (int o = 0; o < 64; ++o)
    v += (float)q0[qo + o] * (float)k0[eo + o] + (float)q1[qo + o] * (float)k1[eo + o] +
         (float)q2[qo + o] * (float)k2[eo + o];
  vl[e] = v / sqrtf(192.f);
}
__global__ void k_segsoftmax(const float* lg, float* w, int nseg, int L, int H) {
  int t = blockIdx.x * blockDim.x + threadIdx.x;
  if (t >= nseg * H) return;
  int seg = t / H, h = t % H;
  size_t base = (size_t)seg * L;
  float m = -3.4e38f;
  for (int j = 0; j < L; ++j) { float v = lg[(base + j) * H + h]; m = v > m ? v : m; }
  if (!(m == m) || fabsf(m) > 3.0e38f) m = 0.f;
  float s = 0.f;
  for (int j = 0; j < L; ++j) s += __expf(lg[(base + j) * H + h] - m);
  float inv = 1.f / (s + 1e-16f);
  for (int j = 0; j < L; ++j) w[(base + j) * H + h] = __expf(lg[(base + j) * H + h] - m) * inv;
}
__global__ void k_segmix(const float* wS, const float* wV, const f16* vs,
                         const f16* v0, const f16* v1, const f16* v2,
                         f16* hs, f16* h0, f16* h1, f16* h2, int L) {
  int q = blockIdx.x, t = threadIdx.x;
  size_t base = (size_t)q * L;
  for (int d = t; d < 256; d += 128) {
    float acc = 0.f;
    for (int j = 0; j < L; ++j) acc += wS[base + j] * (float)vs[(base + j) * 256 + d];
    hs[(size_t)q * 256 + d] = (f16)acc;
  }
  for (int o = t; o < 64; o += 128) {
    float a0 = 0.f, a1 = 0.f, a2 = 0.f;
    for (int j = 0; j < L; ++j) {
      float w = wV[base + j]; size_t ix = (base + j) * 64 + o;
      a0 += w * (float)v0[ix]; a1 += w * (float)v1[ix]; a2 += w * (float)v2[ix];
    }
    h0[(size_t)q * 64 + o] = (f16)a0; h1[(size_t)q * 64 + o] = (f16)a1;
    h2[(size_t)q * 64 + o] = (f16)a2;
  }
}
__global__ void k_copy(const f16* src, int lds, const int* idx, f16* dst, int ldd,
                       int dcol, int cols, long long n) {
  long long i = (long long)blockIdx.x * blockDim.x + threadIdx.x;
  if (i >= n) return;
  int r = (int)(i / cols), c = (int)(i % cols);
  int sr = idx ? idx[r] : r;
  dst[(size_t)r * ldd + dcol + c] = src[(size_t)sr * lds + c];
}
__global__ void k_tri_logits(const int* pi, const int* pj, const float* bs,
                             const f16* b0, const f16* b1, const f16* b2,
                             const f16* qs, const f16* ks,
                             const f16* q0, const f16* q1, const f16* q2,
                             const f16* k0, const f16* k1, const f16* k2,
                             float* tls, float* tlv, long long n) {
  long long t = (long long)blockIdx.x * blockDim.x + threadIdx.x;
  if (t >= n) return;
  int p = (int)(t >> 2), h = (int)(t & 3);
  int i = pi[p], j = pj[p];
  const f16* a = qs + (size_t)i * 256 + h * 64;
  const f16* b = ks + (size_t)j * 256 + h * 64;
  float s = bs[t];
  for (int d = 0; d < 64; ++d) s += (float)a[d] * (float)b[d];
  tls[t] = s;
  float x0 = (float)b0[t], x1 = (float)b1[t], x2 = (float)b2[t];
  float v = x0 * x0 + x1 * x1 + x2 * x2;
  size_t io = (size_t)i * 64 + h * 16, jo = (size_t)j * 64 + h * 16;
  for (int o = 0; o < 16; ++o)
    v += (float)q0[io + o] * (float)k0[jo + o] + (float)q1[io + o] * (float)k1[jo + o] +
         (float)q2[io + o] * (float)k2[jo + o];
  tlv[t] = v;
}
__global__ void k_tri_agg(const float* wS, const float* wV, const int* pj,
                          const f16* vs, const f16* v0, const f16* v1, const f16* v2,
                          float* os, float* o0, float* o1, float* o2) {
  int i = blockIdx.x, t = threadIdx.x, pb = i * 8;
  for (int d = t; d < 256; d += 128) {
    int h = d >> 6; float acc = 0.f;
    for (int b = 0; b < 8; ++b) {
      int p = pb + b; int j = pj[p];
      acc += wS[(size_t)p * 4 + h] * (float)vs[(size_t)j * 256 + d];
    }
    os[(size_t)i * 256 + d] = acc;
  }
  for (int o = t; o < 64; o += 128) {
    int h = o >> 4; float a0 = 0.f, a1 = 0.f, a2 = 0.f;
    for (int b = 0; b < 8; ++b) {
      int p = pb + b; int j = pj[p]; float w = wV[(size_t)p * 4 + h];
      size_t ix = (size_t)j * 64 + o;
      a0 += w * (float)v0[ix]; a1 += w * (float)v1[ix]; a2 += w * (float)v2[ix];
    }
    o0[(size_t)i * 64 + o] = a0; o1[(size_t)i * 64 + o] = a1; o2[(size_t)i * 64 + o] = a2;
  }
}
__device__ __forceinline__ float wred(float x) {
  for (int o = 16; o; o >>= 1) x += __shfl_xor(x, o, 32);
  return x;
}
__global__ void k_ln_sca(const float* x, const float* r, const float* gq, const float* bq,
                         f16* out, int M) {
  int w = (blockIdx.x * blockDim.x + threadIdx.x) >> 5;
  int lane = threadIdx.x & 31;
  if (w >= M) return;
  float v[8]; float s = 0.f;
  for (int k = 0; k < 8; ++k) {
    size_t ix = (size_t)w * 256 + lane * 8 + k;
    v[k] = x[ix] + r[ix]; s += v[k];
  }
  float mean = wred(s) / 256.f;
  float ss = 0.f;
  for (int k = 0; k < 8; ++k) { float t = v[k] - mean; ss += t * t; }
  float var = wred(ss) / 256.f;
  float inv = rsqrtf(var + 1e-5f);
  for (int k = 0; k < 8; ++k) {
    int d = lane * 8 + k;
    out[(size_t)w * 256 + d] = (f16)((v[k] - mean) * inv * gq[d] + bq[d]);
  }
}
__global__ void k_ln_vec(const float* x0, const float* x1, const float* x2,
                         const float* r0, const float* r1, const float* r2,
                         const float* gq, const float* bq,
                         f16* o0, f16* o1, f16* o2, int M) {
  int w = (blockIdx.x * blockDim.x + threadIdx.x) >> 5;
  int lane = threadIdx.x & 31;
  if (w >= M) return;
  float v[6]; float s = 0.f;
  for (int k = 0; k < 2; ++k) {
    size_t ix = (size_t)w * 64 + lane * 2 + k;
    v[k * 3 + 0] = x0[ix] + r0[ix]; v[k * 3 + 1] = x1[ix] + r1[ix]; v[k * 3 + 2] = x2[ix] + r2[ix];
    s += v[k * 3] + v[k * 3 + 1] + v[k * 3 + 2];
  }
  float mean = wred(s) / 192.f;
  float ss = 0.f;
  for (int k = 0; k < 6; ++k) { float t = v[k] - mean; ss += t * t; }
  float var = wred(ss) / 192.f;
  float inv = rsqrtf(var + 1e-5f);
  for (int k = 0; k < 2; ++k) {
    int o = lane * 2 + k; size_t ix = (size_t)w * 64 + o;
    o0[ix] = (f16)((v[k * 3 + 0] - mean) * inv * gq[o * 3 + 0] + bq[o * 3 + 0]);
    o1[ix] = (f16)((v[k * 3 + 1] - mean) * inv * gq[o * 3 + 1] + bq[o * 3 + 1]);
    o2[ix] = (f16)((v[k * 3 + 2] - mean) * inv * gq[o * 3 + 2] + bq[o * 3 + 2]);
  }
}
__global__ void k_flow(const float* pred, float* z, float* lj, int n) {
  int t = blockIdx.x * blockDim.x + threadIdx.x;
  if (t >= n) return;
  int e = t >> 2, j = t & 3;
  float sr = pred[(size_t)e * 8 + j], tt = pred[(size_t)e * 8 + 4 + j];
  float s = __expf(sr);
  z[t] = (z[t] + tt) * s;
  lj[t] += logf(fabsf(s) + 1e-20f);
}
__global__ void k_init(const float* z0, float* z, float* lj, int n) {
  int i = blockIdx.x * blockDim.x + threadIdx.x;
  if (i >= n) return;
  z[i] = z0[i]; lj[i] = 0.f;
}

// ---------------------------------------------------------------------------
// host orchestration
// ---------------------------------------------------------------------------
struct GbW { const f16 *gw, *ls, *lv1, *lv2; const float* gb_; int in_s, in_v, out_s, out_v, hid; };
struct LinW { const float* b; const f16* w; };

extern "C" void kernel_launch(void* const* d_in, const int* in_sizes, int n_in,
                              void* d_out, int out_size, void* d_ws, size_t ws_size,
                              hipStream_t stream) {
  (void)in_sizes; (void)n_in; (void)out_size; (void)ws_size;
  const float* pf   = (const float*)d_in[0];
  const float* z0   = (const float*)d_in[1];
  const float* posq = (const float*)d_in[2];
  const float* posc = (const float*)d_in[3];
  const float* nas  = (const float*)d_in[4];
  const float* navi = (const float*)d_in[5];
  const float* atom = (const float*)d_in[6];
  const int*   eiq  = (const int*)d_in[7];
  const int*   knn  = (const int*)d_in[8];
  const int*   iatt = (const int*)d_in[9];
  const int*   trie = (const int*)d_in[10];
  const float* trif = (const float*)d_in[11];

  const int* qik = knn;            const int* cjk = knn + EKn;
  const int* qi2 = eiq;            const int* cj2 = eiq + EQn;
  const int* pi  = iatt;           const int* pj  = iatt + Pn;
  const int* ta  = trie;           const int* tb  = trie + Pn;

  char* wsb = (char*)d_ws; size_t cur = 0;
  auto alloc = [&](size_t bytes) { cur = (cur + 255) & ~(size_t)255; void* p = wsb + cur; cur += bytes; return p; };
  auto af16 = [&](size_t n) { return (f16*)alloc(n * sizeof(f16)); };
  auto af32 = [&](size_t n) { return (float*)alloc(n * sizeof(float)); };
  auto grid1 = [](long long n) { return dim3((unsigned)((n + 255) / 256)); };

  // ---- parameter pytree walk (jax tree_flatten = sorted dict keys) ----
  size_t po = 0;
  auto ptake = [&](size_t n) { size_t r = po; po += n; return r; };
  auto wf16 = [&](size_t n) {
    f16* d = af16(n);
    k_cvt<<<grid1((long long)n), 256, 0, stream>>>(pf + ptake(n), d, (long long)n);
    return (const f16*)d;
  };
  auto bf32v = [&](size_t n) { return pf + ptake(n); };
  auto mk_gb = [&](int is, int iv, int os, int ov) {
    GbW w; w.in_s = is; w.in_v = iv; w.out_s = os; w.out_v = ov; w.hid = iv > ov ? iv : ov;
    w.gb_ = bf32v(ov);                       // gate.b
    w.gw  = wf16((size_t)ov * os);           // gate.w
    w.ls  = wf16((size_t)os * (is + w.hid)); // ls
    w.lv1 = wf16((size_t)w.hid * iv);        // lv1
    w.lv2 = wf16((size_t)ov * w.hid);        // lv2
    return w;
  };
  struct LayerW {
    GbW bias_gb; const float* bias_vexp; GbW katt;
    const float *ln_s_b, *ln_s_g, *ln_v_b, *ln_v_g;
    GbW qatt, vatt;
    const f16* ef0_dir; GbW ef0; GbW ef1; GbW pred;
    const f16* ne0_dir; GbW ne0; GbW ne1; const float* vexp;
  };
  LayerW L[3];
  for (int i = 0; i < 3; ++i) {           // "layers" sorts before "pos_enc"
    L[i].bias_gb = mk_gb(256, 64, 4, 4);
    L[i].bias_vexp = bf32v(64);
    L[i].katt = mk_gb(256, 64, 256, 64);
    L[i].ln_s_b = bf32v(256); L[i].ln_s_g = bf32v(256);
    L[i].ln_v_b = bf32v(192); L[i].ln_v_g = bf32v(192);
    L[i].qatt = mk_gb(256, 64, 256, 64);
    L[i].vatt = mk_gb(256, 64, 256, 64);
    L[i].ef0_dir = wf16(64 * 64); L[i].ef0 = mk_gb(768, 192, 256, 64);
    L[i].ef1 = mk_gb(256, 64, 256, 64);
    L[i].pred = mk_gb(256, 64, 8, 1);
    L[i].ne0_dir = wf16(64 * 64); L[i].ne0 = mk_gb(64, 64, 256, 64);
    L[i].ne1 = mk_gb(256, 64, 256, 64);
    L[i].vexp = bf32v(64);
  }
  GbW att_k = mk_gb(256, 64, 256, 64);
  GbW att_q = mk_gb(256, 64, 256, 64);
  GbW att_v = mk_gb(256, 64, 256, 64);
  LinW e2n; e2n.b = bf32v(64); e2n.w = wf16((size_t)64 * 64);
  const f16* egbp_dir = wf16(64 * 64); GbW egbp = mk_gb(64, 64, 64, 64);
  const f16* edge_vn = wf16((size_t)64 * 64);
  LinW n2e; n2e.b = bf32v(64); n2e.w = wf16((size_t)64 * 256);
  GbW node_gb = mk_gb(256, 64, 256, 64);
  GbW outgb = mk_gb(256, 64, 256, 64);
  LinW sca_lin; sca_lin.b = bf32v(256); sca_lin.w = wf16((size_t)256 * 64);
  GbW root_lin = mk_gb(256, 64, 256, 64);
  const float* root_vexp = bf32v(64);
  const float* pe_vexp = bf32v(64);

  // ---- persistent activations ----
  f16* atom16 = af16((size_t)NQn * 256);
  k_cvt<<<grid1((long long)NQn * 256), 256, 0, stream>>>(atom, atom16, (long long)NQn * 256);
  f16* nasca16 = af16((size_t)NCn * 256);
  k_cvt<<<grid1((long long)NCn * 256), 256, 0, stream>>>(nas, nasca16, (long long)NCn * 256);
  f16* navp[3]; for (int c = 0; c < 3; ++c) navp[c] = af16((size_t)NCn * 64);
  k_planes<<<grid1((long long)NCn * 64), 256, 0, stream>>>(navi, navp[0], navp[1], navp[2], (long long)NCn * 64);
  f16* hqs = af16((size_t)NQn * 256);
  f16* hqv[3]; for (int c = 0; c < 3; ++c) hqv[c] = af16((size_t)NQn * 64);
  f16* hpS[3]; for (int c = 0; c < 3; ++c) hpS[c] = af16((size_t)Pn * 64);
  f16* hnS = af16((size_t)Pn * 64);
  float* gateS = af32((size_t)EKn * 64);
  float* sca32S = af32((size_t)EQn * 256);

  auto gemm = [&](int M, int N, int K, const f16* A1, int lda1, int K1,
                  const f16* A2, int lda2, const int* idx,
                  const f16* W, int ldw, const float* bias, const float* gate,
                  float* c32, f16* c16, int flags) {
    GemmArgs ga;
    ga.A1 = A1; ga.A2 = A2; ga.W = W; ga.idxA = idx; ga.bias = bias; ga.gate = gate;
    ga.Cf32 = c32; ga.Cf16 = c16; ga.M = M; ga.N = N; ga.K = K;
    ga.K1 = A2 ? K1 : K; ga.lda1 = lda1; ga.lda2 = lda2; ga.ldw = ldw; ga.flags = flags;
    dim3 gr((M + 31) / 32, (N + 63) / 64);
    k_gemm<<<gr, dim3(128), 0, stream>>>(ga);
  };
  auto run_gb = [&](const GbW& g, int M, const f16* sca, f16* const* vp,
                    float* osca32, f16* osca16, f16* const* ov16, float* const* ov32,
                    bool do_vec) {
    for (int c = 0; c < 3; ++c)
      gemm(M, g.hid, g.in_v, vp[c], g.in_v, 0, nullptr, 0, nullptr,
           g.lv1, g.in_v, nullptr, nullptr, nullptr, hpS[c], 0);
    long long hn = (long long)M * g.hid;
    k_vecnorm3<<<grid1(hn), 256, 0, stream>>>(hpS[0], hpS[1], hpS[2], hnS, hn);
    gemm(M, g.out_s, g.hid + g.in_s, hnS, g.hid, g.hid, sca, g.in_s, nullptr,
         g.ls, g.hid + g.in_s, nullptr, nullptr, osca32, osca16, 0);
    if (!do_vec) return;
    gemm(M, g.out_v, g.out_s, osca16, g.out_s, 0, nullptr, 0, nullptr,
         g.gw, g.out_s, g.gb_, nullptr, gateS, nullptr, GF_BIAS | GF_SIGMOID);
    for (int c = 0; c < 3; ++c)
      gemm(M, g.out_v, g.hid, hpS[c], g.hid, 0, nullptr, 0, nullptr,
           g.lv2, g.hid, nullptr, gateS, ov32 ? ov32[c] : nullptr, ov16[c], GF_GATE);
  };
  auto run_gbp = [&](const GbW& g, const f16* dir, int M, const f16* sca, f16* const* vp,
                     float* osca32, f16* osca16, f16* const* ov16) {
    run_gb(g, M, sca, vp, osca32, osca16, ov16, nullptr, true);
    k_leaky<<<grid1((long long)M * g.out_s), 256, 0, stream>>>(osca32, osca16, (long long)M * g.out_s);
    for (int c = 0; c < 3; ++c)
      gemm(M, g.out_v, g.out_v, ov16[c], g.out_v, 0, nullptr, 0, nullptr,
           dir, g.out_v, nullptr, nullptr, nullptr, hpS[c], 0);
    k_vnleaky<<<grid1((long long)M * g.out_v), 256, 0, stream>>>(
        ov16[0], ov16[1], ov16[2], hpS[0], hpS[1], hpS[2], (long long)M * g.out_v);
  };

  // =================== stage 1: positional encoder ===================
  size_t mark = cur;
  f16* esk = af16((size_t)EKn * 64);
  f16* evk[3]; for (int c = 0; c < 3; ++c) evk[c] = af16((size_t)EKn * 64);
  f16* rootvp[3]; for (int c = 0; c < 3; ++c) rootvp[c] = af16((size_t)NQn * 64);
  f16* xs = af16((size_t)NQn * 256);
  f16* xv[3]; for (int c = 0; c < 3; ++c) xv[c] = af16((size_t)NQn * 64);
  f16* ns = af16((size_t)NCn * 256);
  f16* nvp[3]; for (int c = 0; c < 3; ++c) nvp[c] = af16((size_t)NCn * 64);
  f16* ese = af16((size_t)EKn * 64);
  f16* eve[3]; for (int c = 0; c < 3; ++c) eve[c] = af16((size_t)EKn * 64);
  f16* t1 = af16((size_t)EKn * 256);
  f16* ys = af16((size_t)EKn * 256);
  float* cA = af32((size_t)EKn * 64);
  float* cB = af32((size_t)EKn * 64);
  f16* evn[3]; for (int c = 0; c < 3; ++c) evn[c] = af16((size_t)EKn * 64);
  f16* yv[3]; for (int c = 0; c < 3; ++c) yv[c] = af16((size_t)EKn * 64);
  f16* ms = af16((size_t)EKn * 256);
  f16* mv[3]; for (int c = 0; c < 3; ++c) mv[c] = af16((size_t)EKn * 64);
  f16* qs1 = af16((size_t)NQn * 256);
  f16* qv1[3]; for (int c = 0; c < 3; ++c) qv1[c] = af16((size_t)NQn * 64);
  f16* ks1 = af16((size_t)EKn * 256);
  f16* kv1[3]; for (int c = 0; c < 3; ++c) kv1[c] = af16((size_t)EKn * 64);
  f16* vs1 = af16((size_t)EKn * 256);
  f16* vv1[3]; for (int c = 0; c < 3; ++c) vv1[c] = af16((size_t)EKn * 64);
  float* sl = af32(EKn); float* vlg = af32(EKn);
  float* swt = af32(EKn); float* vwt = af32(EKn);

  k_edge_geom<<<grid1(EKn), 256, 0, stream>>>(posq, posc, qik, cjk, EKn, pe_vexp, 64,
      esk, 64, 10.f, 64, nullptr, 0, evk[0], evk[1], evk[2], 64);
  k_edge_geom<<<grid1(NQn), 256, 0, stream>>>(posq, nullptr, nullptr, nullptr, NQn, root_vexp, 64,
      nullptr, 0, 0.f, 0, nullptr, 0, rootvp[0], rootvp[1], rootvp[2], 64);
  run_gb(root_lin, NQn, atom16, rootvp, nullptr, xs, xv, nullptr, true);
  run_gb(node_gb, NCn, nasca16, navp, nullptr, ns, nvp, nullptr, true);
  run_gbp(egbp, egbp_dir, EKn, esk, evk, sca32S, ese, eve);
  gemm(EKn, 256, 64, ese, 64, 0, nullptr, 0, nullptr, sca_lin.w, 64, sca_lin.b, nullptr,
       nullptr, t1, GF_BIAS);
  k_mulg<<<grid1((long long)EKn * 256), 256, 0, stream>>>(ns, cjk, t1, ys, 256, (long long)EKn * 256);
  gemm(EKn, 64, 64, ese, 64, 0, nullptr, 0, nullptr, e2n.w, 64, e2n.b, nullptr, cA, nullptr, GF_BIAS);
  gemm(EKn, 64, 256, ns, 256, 0, nullptr, 0, cjk, n2e.w, 256, n2e.b, nullptr, cB, nullptr, GF_BIAS);
  for (int c = 0; c < 3; ++c)
    gemm(EKn, 64, 64, eve[c], 64, 0, nullptr, 0, nullptr, edge_vn, 64, nullptr, nullptr,
         nullptr, evn[c], 0);
  k_yvec<<<grid1((long long)EKn * 64), 256, 0, stream>>>(cA, cB, cjk,
      nvp[0], nvp[1], nvp[2], evn[0], evn[1], evn[2], yv[0], yv[1], yv[2], (long long)EKn * 64);
  run_gb(outgb, EKn, ys, yv, nullptr, ms, mv, nullptr, true);
  run_gb(att_q, NQn, xs, xv, nullptr, qs1, qv1, nullptr, true);
  run_gb(att_k, EKn, ms, mv, nullptr, ks1, kv1, nullptr, true);
  run_gb(att_v, EKn, ms, mv, nullptr, vs1, vv1, nullptr, true);
  k_att1<<<grid1(EKn), 256, 0, stream>>>(qik, qs1, ks1, qv1[0], qv1[1], qv1[2],
      kv1[0], kv1[1], kv1[2], sl, vlg, EKn);
  k_segsoftmax<<<grid1(NQn), 256, 0, stream>>>(sl, swt, NQn, KKNN, 1);
  k_segsoftmax<<<grid1(NQn), 256, 0, stream>>>(vlg, vwt, NQn, KKNN, 1);
  k_segmix<<<NQn, 128, 0, stream>>>(swt, vwt, vs1, vv1[0], vv1[1], vv1[2],
      hqs, hqv[0], hqv[1], hqv[2], KKNN);

  // =================== stage 2: flow layers (reuse stage-1 temp space) ===================
  cur = mark;
  f16* fsin = af16((size_t)EQn * 64);
  f16* fvin[3]; for (int c = 0; c < 3; ++c) fvin[c] = af16((size_t)EQn * 64);
  f16* fs0 = af16((size_t)EQn * 256);
  f16* fv0[3]; for (int c = 0; c < 3; ++c) fv0[c] = af16((size_t)EQn * 64);
  f16* fs1 = af16((size_t)EQn * 256);
  f16* fv1[3]; for (int c = 0; c < 3; ++c) fv1[c] = af16((size_t)EQn * 64);
  f16* esin = af16((size_t)EQn * 768);
  f16* evin[3]; for (int c = 0; c < 3; ++c) evin[c] = af16((size_t)EQn * 192);
  f16* es0 = af16((size_t)EQn * 256);
  f16* ev0[3]; for (int c = 0; c < 3; ++c) ev0[c] = af16((size_t)EQn * 64);
  float* es32 = af32((size_t)EQn * 256);
  f16* es16 = af16((size_t)EQn * 256);
  float* ev32[3]; for (int c = 0; c < 3; ++c) ev32[c] = af32((size_t)EQn * 64);
  f16* ev16[3]; for (int c = 0; c < 3; ++c) ev16[c] = af16((size_t)EQn * 64);
  f16* qs2 = af16((size_t)EQn * 256);
  f16* qv2[3]; for (int c = 0; c < 3; ++c) qv2[c] = af16((size_t)EQn * 64);
  f16* ks2 = af16((size_t)EQn * 256);
  f16* kv2[3]; for (int c = 0; c < 3; ++c) kv2[c] = af16((size_t)EQn * 64);
  f16* vs2 = af16((size_t)EQn * 256);
  f16* vv2[3]; for (int c = 0; c < 3; ++c) vv2[c] = af16((size_t)EQn * 64);
  f16* tsca = af16((size_t)Pn * 256);
  f16* tvp[3]; for (int c = 0; c < 3; ++c) tvp[c] = af16((size_t)Pn * 64);
  float* bs32 = af32((size_t)Pn * 4);
  f16* bs16 = af16((size_t)Pn * 4);
  f16* bvp[3]; for (int c = 0; c < 3; ++c) bvp[c] = af16((size_t)Pn * 4);
  float* tls = af32((size_t)Pn * 4); float* tlv = af32((size_t)Pn * 4);
  float* twS = af32((size_t)Pn * 4); float* twV = af32((size_t)Pn * 4);
  float* ats = af32((size_t)EQn * 256);
  float* atv[3]; for (int c = 0; c < 3; ++c) atv[c] = af32((size_t)EQn * 64);
  f16* esb = af16((size_t)EQn * 256);
  f16* evb[3]; for (int c = 0; c < 3; ++c) evb[c] = af16((size_t)EQn * 64);
  float* ps = af32((size_t)EQn * 8);

  float* zo = (float*)d_out;
  float* lj = (float*)d_out + (size_t)EQn * 4;
  k_init<<<grid1((long long)EQn * 4), 256, 0, stream>>>(z0, zo, lj, EQn * 4);

  for (int li = 0; li < 3; ++li) {
    const LayerW& lw = L[li];
    k_edge_geom<<<grid1(EQn), 256, 0, stream>>>(posq, posc, qi2, cj2, EQn, lw.vexp, 64,
        fsin, 64, 3.f, 64, nullptr, 0, fvin[0], fvin[1], fvin[2], 64);
    run_gbp(lw.ne0, lw.ne0_dir, EQn, fsin, fvin, sca32S, fs0, fv0);
    run_gb(lw.ne1, EQn, fs0, fv0, nullptr, fs1, fv1, nullptr, true);
    // concat [h_sca[qi], na_sca[cj], fs] / [h_vec[qi], na_vec[cj], fv]
    k_copy<<<grid1((long long)EQn * 256), 256, 0, stream>>>(hqs, 256, qi2, esin, 768, 0, 256, (long long)EQn * 256);
    k_copy<<<grid1((long long)EQn * 256), 256, 0, stream>>>(nasca16, 256, cj2, esin, 768, 256, 256, (long long)EQn * 256);
    k_copy<<<grid1((long long)EQn * 256), 256, 0, stream>>>(fs1, 256, nullptr, esin, 768, 512, 256, (long long)EQn * 256);
    for (int c = 0; c < 3; ++c) {
      k_copy<<<grid1((long long)EQn * 64), 256, 0, stream>>>(hqv[c], 64, qi2, evin[c], 192, 0, 64, (long long)EQn * 64);
      k_copy<<<grid1((long long)EQn * 64), 256, 0, stream>>>(navp[c], 64, cj2, evin[c], 192, 64, 64, (long long)EQn * 64);
      k_copy<<<grid1((long long)EQn * 64), 256, 0, stream>>>(fv1[c], 64, nullptr, evin[c], 192, 128, 64, (long long)EQn * 64);
    }
    run_gbp(lw.ef0, lw.ef0_dir, EQn, esin, evin, sca32S, es0, ev0);
    run_gb(lw.ef1, EQn, es0, ev0, es32, es16, ev16, ev32, true);
    run_gb(lw.qatt, EQn, es16, ev16, nullptr, qs2, qv2, nullptr, true);
    run_gb(lw.katt, EQn, es16, ev16, nullptr, ks2, kv2, nullptr, true);
    run_gb(lw.vatt, EQn, es16, ev16, nullptr, vs2, vv2, nullptr, true);
    // triangle-edge attention bias
    k_edge_geom<<<grid1(Pn), 256, 0, stream>>>(posc, posc, ta, tb, Pn, lw.bias_vexp, 64,
        tsca, 256, 10.f, 252, trif, 4, tvp[0], tvp[1], tvp[2], 64);
    run_gb(lw.bias_gb, Pn, tsca, tvp, bs32, bs16, bvp, nullptr, true);
    k_tri_logits<<<grid1((long long)Pn * 4), 256, 0, stream>>>(pi, pj, bs32,
        bvp[0], bvp[1], bvp[2], qs2, ks2, qv2[0], qv2[1], qv2[2], kv2[0], kv2[1], kv2[2],
        tls, tlv, (long long)Pn * 4);
    k_segsoftmax<<<grid1((long long)EQn * 4), 256, 0, stream>>>(tls, twS, EQn, DEGQ, 4);
    k_segsoftmax<<<grid1((long long)EQn * 4), 256, 0, stream>>>(tlv, twV, EQn, DEGQ, 4);
    k_tri_agg<<<EQn, 128, 0, stream>>>(twS, twV, pj, vs2, vv2[0], vv2[1], vv2[2],
        ats, atv[0], atv[1], atv[2]);
    k_ln_sca<<<grid1((long long)EQn * 32), 256, 0, stream>>>(es32, ats, lw.ln_s_g, lw.ln_s_b, esb, EQn);
    k_ln_vec<<<grid1((long long)EQn * 32), 256, 0, stream>>>(ev32[0], ev32[1], ev32[2],
        atv[0], atv[1], atv[2], lw.ln_v_g, lw.ln_v_b, evb[0], evb[1], evb[2], EQn);
    // edge_pred: only scalar head needed
    run_gb(lw.pred, EQn, esb, evb, ps, nullptr, nullptr, nullptr, false);
    k_flow<<<grid1((long long)EQn * 4), 256, 0, stream>>>(ps, zo, lj, EQn * 4);
  }
}
